// dressed_quantum_circuit_29832842838612
// MI455X (gfx1250) — matrix operand order repaired and checked by reading, not compiled, not run on hardware
//
#include <hip/hip_runtime.h>
#include <hip/hip_bf16.h>
#include <math.h>

typedef __attribute__((ext_vector_type(16))) _Float16 v16h;
typedef __attribute__((ext_vector_type(8)))  float    v8f;

#define NQ 10
#define QD 6
#define BATCH 8192
#define DIN 512
#define DOUT 64

// ---------------- workspace layout ----------------
// [0,      16384)  : W1 fragment, 16 k-steps x 32 lanes x 16 halves (B-layout)
// [16384,  16864)  : qcs, 60 float2 (cos,sin of half q_weights)
// [16896, 672256)  : cs, 8192*10 float2 (cos,sin of half input angles)

__device__ __forceinline__ float shx(float v, int m) { return __shfl_xor(v, m, 32); }

// ---------------- kernel 0: prep ----------------
__global__ void prep_kernel(const float* __restrict__ W1,
                            const float* __restrict__ qw,
                            _Float16* __restrict__ w1frag,
                            float2* __restrict__ qcs) {
  int idx = blockIdx.x * 256 + threadIdx.x;
  if (idx < 16 * 512) {
    // B fragment for v_wmma_f32_16x16x32_f16:
    // half j<8 -> K=hi*8+j, j>=8 -> K=16+hi*8+(j-8) (hi = lane>>4); N = lane&15
    int kk = idx >> 9;
    int rem = idx & 511;
    int l = rem >> 4;
    int j = rem & 15;
    int K = kk * 32 + ((j >> 3) << 4) + ((l >> 4) << 3) + (j & 7);
    int n = l & 15;
    float v = (n < NQ) ? W1[K * NQ + n] : 0.0f;
    w1frag[idx] = (_Float16)v;
  } else if (idx < 16 * 512 + QD * NQ) {
    int t = idx - 16 * 512;
    float s, c;
    __sincosf(qw[t] * 0.5f, &s, &c);
    qcs[t] = make_float2(c, s);
  }
}

// ---------------- kernel 1: x@W1 via WMMA -> (cos,sin) of half-angles ----------------
// 8 waves per 16-row tile, each covering K=64 (2 WMMA steps); LDS tree-combines the
// partial accumulators. 512 blocks * 8 waves = 4096 waves -> enough loads in flight
// to approach HBM peak on the 16.8 MB read of x.
__global__ void __launch_bounds__(256)
gemm1_kernel(const float* __restrict__ x,
             const float* __restrict__ b1,
             const _Float16* __restrict__ w1frag,
             float2* __restrict__ cs) {
  __shared__ float red[7 * 256 * 8];  // 7 waves x 32 lanes x 8 floats = 7 KB

  int tile = blockIdx.x;              // row tile 0..511
  int wv   = threadIdx.x >> 5;        // 0..7
  int lane = threadIdx.x & 31;
  int m  = lane & 15;
  int hi = lane >> 4;
  const float* xrow = x + (size_t)(tile * 16 + m) * DIN;

  v8f acc = {};
  #pragma unroll
  for (int t = 0; t < 2; ++t) {
    int kk = wv * 2 + t;
    // A fragment (16x32 f16): lanes 0-15 rows, halves 0-7 -> K=hi*8+j, 8-15 -> K=16+hi*8+j
    int kbase = kk * 32 + hi * 8;
    v16h a;
    #pragma unroll
    for (int j = 0; j < 8; ++j) a[j]     = (_Float16)xrow[kbase + j];
    #pragma unroll
    for (int j = 0; j < 8; ++j) a[8 + j] = (_Float16)xrow[kbase + 16 + j];
    v16h b = *(const v16h*)(w1frag + kk * 512 + lane * 16);
    acc = __builtin_amdgcn_wmma_f32_16x16x32_f16(false, a, false, b,
                                                 (short)0, acc, false, false);
  }

  if (wv > 0) {
    #pragma unroll
    for (int r = 0; r < 8; ++r) red[(wv - 1) * 256 + lane * 8 + r] = acc[r];
  }
  __syncthreads();
  if (wv == 0) {
    #pragma unroll
    for (int w2 = 0; w2 < 7; ++w2)
      #pragma unroll
      for (int r = 0; r < 8; ++r) acc[r] += red[w2 * 256 + lane * 8 + r];
    // C layout: VGPR r -> M = r + hi*8, N = lane&15
    if (m < NQ) {
      float bb = b1[m];
      #pragma unroll
      for (int r = 0; r < 8; ++r) {
        int row = tile * 16 + r + hi * 8;
        float ang = tanhf(acc[r] + bb) * 1.5707963267948966f;
        float s, c;
        __sincosf(ang * 0.5f, &s, &c);
        cs[row * NQ + m] = make_float2(c, s);
      }
    }
  }
}

// ---------------- VQC helpers ----------------
// Wire -> bit assignment (alternating, minimizes CNOT shuffle cost):
//   even wires 0,2,4,6,8 -> LANE masks 16,8,4,2,1
//   odd  wires 1,3,5,7,9 -> REG  masks 16,8,4,2,1
// Entangler ctrl is always the lower wire, so:
//   even brick (0,1)(2,3)(4,5)(6,7)(8,9) = lane-ctrl -> reg-tgt : pure v_cndmask, NO shuffles
//   odd  brick (1,2)(3,4)(5,6)(7,8)      = reg-ctrl -> lane-tgt : 16 shuffles each

template <int STR>  // RY on a register wire: pair stride STR in register index
__device__ __forceinline__ void ry_reg(float (&st)[32], float c, float s) {
  #pragma unroll
  for (int r = 0; r < 32; ++r)
    if ((r & STR) == 0) {
      float a0 = st[r], a1 = st[r | STR];
      st[r]       = c * a0 - s * a1;
      st[r | STR] = s * a0 + c * a1;
    }
}

template <int LM>  // RY on a lane wire: lane mask LM
__device__ __forceinline__ void ry_lane(float (&st)[32], int lane, float c, float s) {
  float s_eff = ((lane & LM) != 0) ? s : -s;   // sign folded once per wire
  #pragma unroll
  for (int r = 0; r < 32; ++r) {
    float p = shx(st[r], LM);
    st[r] = fmaf(s_eff, p, c * st[r]);
  }
}

template <int LCM, int TM>  // CNOT: ctrl = lane bit LCM, tgt = reg bit TM (shuffle-free)
__device__ __forceinline__ void cnot_lr(float (&st)[32], int lane) {
  bool c1 = (lane & LCM) != 0;
  #pragma unroll
  for (int r = 0; r < 32; ++r)
    if ((r & TM) == 0) {
      float a0 = st[r], a1 = st[r | TM];
      st[r]      = c1 ? a1 : a0;
      st[r | TM] = c1 ? a0 : a1;
    }
}

template <int CM, int LTM>  // CNOT: ctrl = reg bit CM, tgt = lane bit LTM (16 shuffles)
__device__ __forceinline__ void cnot_rl(float (&st)[32]) {
  #pragma unroll
  for (int r = 0; r < 32; ++r)
    if (r & CM) st[r] = shx(st[r], LTM);
}

__device__ __forceinline__ float wsum(float v) {
  v += shx(v, 1); v += shx(v, 2); v += shx(v, 4); v += shx(v, 8); v += shx(v, 16);
  return v;
}

// ---------------- kernel 2: VQC (state in 32 VGPRs/lane) + fused @W2+b2 ----------------
__global__ void __launch_bounds__(256)
vqc_kernel(const float2* __restrict__ cs,
           const float2* __restrict__ qcs,
           const float* __restrict__ W2,
           const float* __restrict__ b2,
           float* __restrict__ out) {
  int lane = threadIdx.x & 31;
  int sample = blockIdx.x * (blockDim.x >> 5) + (threadIdx.x >> 5);

  float st[32];
  #pragma unroll
  for (int r = 0; r < 32; ++r) st[r] = 0.03125f;  // 2^(-10/2): H-layer on |0..0>

  // input RY layer (per-sample angles), wire->bit per the alternating map
  const float2* mycs = cs + sample * NQ;
  { float2 p = mycs[0]; ry_lane<16>(st, lane, p.x, p.y); }
  { float2 p = mycs[1]; ry_reg <16>(st, p.x, p.y); }
  { float2 p = mycs[2]; ry_lane< 8>(st, lane, p.x, p.y); }
  { float2 p = mycs[3]; ry_reg < 8>(st, p.x, p.y); }
  { float2 p = mycs[4]; ry_lane< 4>(st, lane, p.x, p.y); }
  { float2 p = mycs[5]; ry_reg < 4>(st, p.x, p.y); }
  { float2 p = mycs[6]; ry_lane< 2>(st, lane, p.x, p.y); }
  { float2 p = mycs[7]; ry_reg < 2>(st, p.x, p.y); }
  { float2 p = mycs[8]; ry_lane< 1>(st, lane, p.x, p.y); }
  { float2 p = mycs[9]; ry_reg < 1>(st, p.x, p.y); }

  // entangler + trainable RY, QD layers (runtime loop keeps I$ footprint small)
  #pragma unroll 1
  for (int k = 0; k < QD; ++k) {
    // even brick: (0,1)(2,3)(4,5)(6,7)(8,9) -- all lane->reg, shuffle-free
    cnot_lr<16, 16>(st, lane);
    cnot_lr< 8,  8>(st, lane);
    cnot_lr< 4,  4>(st, lane);
    cnot_lr< 2,  2>(st, lane);
    cnot_lr< 1,  1>(st, lane);
    // odd brick: (1,2)(3,4)(5,6)(7,8) -- reg->lane, 16 shuffles each
    cnot_rl<16, 8>(st);
    cnot_rl< 8, 4>(st);
    cnot_rl< 4, 2>(st);
    cnot_rl< 2, 1>(st);
    // trainable RY layer
    const float2* q = qcs + k * NQ;
    { float2 p = q[0]; ry_lane<16>(st, lane, p.x, p.y); }
    { float2 p = q[1]; ry_reg <16>(st, p.x, p.y); }
    { float2 p = q[2]; ry_lane< 8>(st, lane, p.x, p.y); }
    { float2 p = q[3]; ry_reg < 8>(st, p.x, p.y); }
    { float2 p = q[4]; ry_lane< 4>(st, lane, p.x, p.y); }
    { float2 p = q[5]; ry_reg < 4>(st, p.x, p.y); }
    { float2 p = q[6]; ry_lane< 2>(st, lane, p.x, p.y); }
    { float2 p = q[7]; ry_reg < 2>(st, p.x, p.y); }
    { float2 p = q[8]; ry_lane< 1>(st, lane, p.x, p.y); }
    { float2 p = q[9]; ry_reg < 1>(st, p.x, p.y); }
  }

  // readout: <Z_w> = sum probs*(+1 if wire bit==0 else -1)
  // odd wires (reg bits): per-register signed sums; even wires (lane bits): signed wsum of tot
  float p1 = 0.f, p3 = 0.f, p5 = 0.f, p7 = 0.f, p9 = 0.f, tot = 0.f;
  #pragma unroll
  for (int r = 0; r < 32; ++r) {
    float pr = st[r] * st[r];
    tot += pr;
    p1 += (r & 16) ? -pr : pr;
    p3 += (r &  8) ? -pr : pr;
    p5 += (r &  4) ? -pr : pr;
    p7 += (r &  2) ? -pr : pr;
    p9 += (r &  1) ? -pr : pr;
  }
  float exps[NQ];
  exps[0] = wsum((lane & 16) ? -tot : tot);
  exps[1] = wsum(p1);
  exps[2] = wsum((lane &  8) ? -tot : tot);
  exps[3] = wsum(p3);
  exps[4] = wsum((lane &  4) ? -tot : tot);
  exps[5] = wsum(p5);
  exps[6] = wsum((lane &  2) ? -tot : tot);
  exps[7] = wsum(p7);
  exps[8] = wsum((lane &  1) ? -tot : tot);
  exps[9] = wsum(p9);

  // fused q_out @ W2 + b2 (K=10, 2 output cols per lane)
  float a0 = b2[lane], a1 = b2[lane + 32];
  #pragma unroll
  for (int w = 0; w < NQ; ++w) {
    a0 = fmaf(exps[w], W2[w * DOUT + lane],      a0);
    a1 = fmaf(exps[w], W2[w * DOUT + lane + 32], a1);
  }
  out[(size_t)sample * DOUT + lane]      = a0;
  out[(size_t)sample * DOUT + lane + 32] = a1;
}

extern "C" void kernel_launch(void* const* d_in, const int* in_sizes, int n_in,
                              void* d_out, int out_size, void* d_ws, size_t ws_size,
                              hipStream_t stream) {
  const float* x  = (const float*)d_in[0];
  const float* W1 = (const float*)d_in[1];
  const float* b1 = (const float*)d_in[2];
  const float* qw = (const float*)d_in[3];
  const float* W2 = (const float*)d_in[4];
  const float* b2 = (const float*)d_in[5];
  float* out = (float*)d_out;

  char* ws = (char*)d_ws;
  _Float16* w1frag = (_Float16*)ws;             // 16384 B
  float2*   qcs    = (float2*)(ws + 16384);     // 480 B
  float2*   cs     = (float2*)(ws + 16896);     // 655360 B

  prep_kernel<<<33, 256, 0, stream>>>(W1, qw, w1frag, qcs);
  gemm1_kernel<<<512, 256, 0, stream>>>(x, b1, w1frag, cs);   // 8 K-split waves per tile
  vqc_kernel<<<BATCH / 8, 256, 0, stream>>>(cs, qcs, W2, b2, out); // 1 wave / sample
}
